// KimiLinearAttention_32066225832166
// MI455X (gfx1250) — matrix operand order repaired and checked by reading, not compiled, not run on hardware
//
#include <hip/hip_runtime.h>
#include <hip/hip_bf16.h>
#include <stdint.h>

#define D_MODEL 2048
#define T_LEN   4096
#define B_SZ    2
#define NHEAD   16
#define DKH     128
#define CHUNK   64
#define M_ROWS  (B_SZ * T_LEN)   // 8192

typedef __attribute__((ext_vector_type(16))) __bf16 v16bf;
typedef __attribute__((ext_vector_type(8)))  __bf16 bf16x8;
typedef __attribute__((ext_vector_type(8)))  float  v8f;

__device__ __forceinline__ __bf16 f2bf(float f) {
    union { float f; uint32_t u; } c; c.f = f;
    uint32_t r = c.u + 0x7FFFu + ((c.u >> 16) & 1u);
    union { uint16_t s; __bf16 b; } o; o.s = (uint16_t)(r >> 16);
    return o.b;
}
__device__ __forceinline__ float bf2f(__bf16 b) {
    union { __bf16 b; uint16_t s; } i; i.b = b;
    union { uint32_t u; float f; } o; o.u = (uint32_t)i.s << 16;
    return o.f;
}
__device__ __forceinline__ v8f zero8() {
    v8f z;
#pragma unroll
    for (int i = 0; i < 8; ++i) z[i] = 0.0f;
    return z;
}
__device__ __forceinline__ v16bf ld16(const __bf16* lo, const __bf16* hi) {
    bf16x8 a = *(const bf16x8*)lo;
    bf16x8 b = *(const bf16x8*)hi;
    v16bf r;
#pragma unroll
    for (int i = 0; i < 8; ++i) { r[i] = a[i]; r[i + 8] = b[i]; }
    return r;
}
__device__ __forceinline__ v8f wmma_bf16(v16bf a, v16bf b, v8f c) {
    return __builtin_amdgcn_wmma_f32_16x16x32_bf16(false, a, false, b, (short)0, c,
                                                   false, false);
}

// ---------------------------------------------------------------- convert
__global__ __launch_bounds__(256) void cvt_f32_bf16(const float* __restrict__ s,
                                                    __bf16* __restrict__ d, int n) {
    int i = (blockIdx.x * 256 + threadIdx.x) * 4;
    if (i < n) {
        float4 v = *(const float4*)(s + i);
        d[i + 0] = f2bf(v.x);
        d[i + 1] = f2bf(v.y);
        d[i + 2] = f2bf(v.z);
        d[i + 3] = f2bf(v.w);
    }
}

// ---------------------------------------------------------------- GEMM
// C[M,2048] = A_bf16[M,2048] @ W_bf16[2048,2048] + bias, compile-time epilogue:
// MODE 0: f32 out; 1: sigmoid f32; 2: bf16 out; 3: RoPE + elu + 1 -> bf16
#define SA_STR 40   // 80 B row stride (16B aligned)
#define M_TILE 256  // 8 waves x 32 rows

template <int MODE>
__global__ __launch_bounds__(256) void gemm_bf16(const __bf16* __restrict__ A,
                                                 const __bf16* __restrict__ W,
                                                 const float* __restrict__ bias,
                                                 void* __restrict__ out) {
    __shared__ __attribute__((aligned(16))) __bf16 sA[M_TILE * SA_STR];  // 20 KB
    __shared__ __attribute__((aligned(16))) __bf16 sB[128 * SA_STR];     // 10 KB

    const int tid = threadIdx.x;
    const int w = tid >> 5, L = tid & 31;
    const int l15 = L & 15, lh = L >> 4;
    const int m0 = blockIdx.y * M_TILE, n0 = blockIdx.x * 128;

    v8f acc[2][8];
#pragma unroll
    for (int g = 0; g < 2; ++g)
#pragma unroll
        for (int j = 0; j < 8; ++j) acc[g][j] = zero8();

    const int bkk = tid >> 3, bnc = (tid & 7) * 16;
    const __bf16* Ap = A + (size_t)(m0 + tid) * D_MODEL;       // one row per thread
    const __bf16* Wp = W + (size_t)bkk * D_MODEL + n0 + bnc;

    for (int k0 = 0; k0 < D_MODEL; k0 += 32) {
        bf16x8 a0 = *(const bf16x8*)(Ap + k0);
        bf16x8 a1 = *(const bf16x8*)(Ap + k0 + 8);
        bf16x8 a2 = *(const bf16x8*)(Ap + k0 + 16);
        bf16x8 a3 = *(const bf16x8*)(Ap + k0 + 24);
        bf16x8 w0 = *(const bf16x8*)(Wp + (size_t)k0 * D_MODEL);
        bf16x8 w1 = *(const bf16x8*)(Wp + (size_t)k0 * D_MODEL + 8);
        __syncthreads();  // previous iteration fragment reads done
        *(bf16x8*)&sA[tid * SA_STR + 0]  = a0;
        *(bf16x8*)&sA[tid * SA_STR + 8]  = a1;
        *(bf16x8*)&sA[tid * SA_STR + 16] = a2;
        *(bf16x8*)&sA[tid * SA_STR + 24] = a3;
#pragma unroll
        for (int m = 0; m < 8; ++m) {
            sB[(bnc + m)     * SA_STR + bkk] = w0[m];
            sB[(bnc + 8 + m) * SA_STR + bkk] = w1[m];
        }
        __syncthreads();
        const __bf16* ab0 = &sA[(32 * w + l15) * SA_STR];
        const __bf16* ab1 = &sA[(32 * w + 16 + l15) * SA_STR];
        v16bf af0 = ld16(ab0 + lh * 16, ab0 + lh * 16 + 8);
        v16bf af1 = ld16(ab1 + lh * 16, ab1 + lh * 16 + 8);
#pragma unroll
        for (int j = 0; j < 8; ++j) {
            const __bf16* bb = &sB[(16 * j + l15) * SA_STR + lh * 16];
            v16bf bf_ = ld16(bb, bb + 8);
            acc[0][j] = wmma_bf16(af0, bf_, acc[0][j]);
            acc[1][j] = wmma_bf16(af1, bf_, acc[1][j]);
        }
    }

    if (MODE == 3) {  // RoPE + elu + 1 (per-head pairs j <-> j+4 live in same lane)
        __bf16* ob = (__bf16*)out;
#pragma unroll
        for (int g = 0; g < 2; ++g) {
#pragma unroll
            for (int j = 0; j < 4; ++j) {
#pragma unroll
                for (int r = 0; r < 8; ++r) {
                    int gm = m0 + 32 * w + 16 * g + r + 8 * lh;
                    int t = gm & (T_LEN - 1);
                    int n1 = n0 + 16 * j + l15;
                    int n2 = n1 + 64;
                    int jj = n1 & (DKH - 1);  // < 64
                    float x1 = acc[g][j][r] + bias[n1];
                    float x2 = acc[g][j + 4][r] + bias[n2];
                    // inv_freq = 10000^(-jj/64) = exp(-ln(1e4)/64 * jj)
                    float ang = (float)t * __expf(-0.14391157f * (float)jj);
                    ang -= 6.2831853f * floorf(ang * 0.15915494f);  // mod 2pi
                    float sn = __sinf(ang), cs = __cosf(ang);
                    float r1 = x1 * cs - x2 * sn;
                    float r2 = x1 * sn + x2 * cs;
                    r1 = r1 > 0.0f ? r1 + 1.0f : __expf(r1);  // elu(x)+1
                    r2 = r2 > 0.0f ? r2 + 1.0f : __expf(r2);
                    ob[(size_t)gm * D_MODEL + n1] = f2bf(r1);
                    ob[(size_t)gm * D_MODEL + n2] = f2bf(r2);
                }
            }
        }
    } else {
#pragma unroll
        for (int g = 0; g < 2; ++g) {
#pragma unroll
            for (int j = 0; j < 8; ++j) {
#pragma unroll
                for (int r = 0; r < 8; ++r) {
                    int gm = m0 + 32 * w + 16 * g + r + 8 * lh;
                    int gn = n0 + 16 * j + l15;
                    float v = acc[g][j][r] + bias[gn];
                    if (MODE == 0)
                        ((float*)out)[(size_t)gm * D_MODEL + gn] = v;
                    else if (MODE == 1)
                        ((float*)out)[(size_t)gm * D_MODEL + gn] =
                            1.0f / (1.0f + __expf(-v));
                    else
                        ((__bf16*)out)[(size_t)gm * D_MODEL + gn] = f2bf(v);
                }
            }
        }
    }
}

// ---------------------------------------------------------------- attention
#define KT_STR 72    // [d|e][i] 64+pad, 144 B rows
#define QS_STR 136   // [i][d] 128+pad, 272 B rows
#define ST_STR 136   // [e][d]
#define ATTN_SMEM ((128 * KT_STR * 2 + 64 * QS_STR + 128 * ST_STR) * 2 + (128 + 256 + 64) * 4)

__global__ __launch_bounds__(256) void attn_kernel(const __bf16* __restrict__ Q,
                                                   const __bf16* __restrict__ K,
                                                   const __bf16* __restrict__ V,
                                                   const float* __restrict__ G,
                                                   __bf16* __restrict__ O) {
    extern __shared__ __attribute__((aligned(16))) char smem[];
    __bf16* kT = (__bf16*)smem;            // 128 x 72
    __bf16* vT = kT + 128 * KT_STR;        // 128 x 72
    __bf16* qS = vT + 128 * KT_STR;        // 64 x 136
    __bf16* sT = qS + 64 * QS_STR;         // 128 x 136 (scum^T in bf16)
    float* zc = (float*)(sT + 128 * ST_STR);  // 128
    float* dp = zc + 128;                  // 64 x 4
    float* dn = dp + 256;                  // 64

    const int tid = threadIdx.x;
    const int w = tid >> 5, L = tid & 31;
    const int l15 = L & 15, lh = L >> 4;
    const int b = blockIdx.x >> 4, h = blockIdx.x & 15;
    const size_t colbase = (size_t)h * DKH;
    const int rw = w & 3;          // num row tile (i)
    const int jt0 = (w >> 2) * 4;  // num col tile base (e)

    v8f scum[8];                   // persistent kv-cumsum: wave owns d in [16w,16w+16)
#pragma unroll
    for (int j = 0; j < 8; ++j) scum[j] = zero8();
    if (tid < 128) zc[tid] = 0.0f;
    __syncthreads();

    for (int n = 0; n < T_LEN / CHUNK; ++n) {
        const size_t rowbase = (size_t)b * T_LEN + n * CHUNK;
        // stage q row-major, k/v transposed
#pragma unroll
        for (int l = 0; l < 4; ++l) {
            int e = tid + l * 256;
            int i = e >> 4, cg = e & 15;
            size_t go = (rowbase + i) * D_MODEL + colbase + cg * 8;
            bf16x8 kv8 = *(const bf16x8*)(K + go);
            bf16x8 vv8 = *(const bf16x8*)(V + go);
            bf16x8 qv8 = *(const bf16x8*)(Q + go);
#pragma unroll
            for (int m = 0; m < 8; ++m) {
                kT[(cg * 8 + m) * KT_STR + i] = kv8[m];
                vT[(cg * 8 + m) * KT_STR + i] = vv8[m];
            }
            *(bf16x8*)&qS[i * QS_STR + cg * 8] = qv8;
        }
        __syncthreads();

        if (tid < 128) {  // zcum += column-sums of k chunk
            float s = 0.0f;
#pragma unroll 8
            for (int i = 0; i < CHUNK; ++i) s += bf2f(kT[tid * KT_STR + i]);
            zc[tid] += s;
        }

        // scum += k^T v  (M=d 128, K=i 64, N=e 128)
#pragma unroll
        for (int s = 0; s < 2; ++s) {
            int kb = 32 * s;
            const __bf16* ab = &kT[(16 * w + l15) * KT_STR + kb];
            v16bf af = ld16(ab + lh * 16, ab + lh * 16 + 8);
#pragma unroll
            for (int j = 0; j < 8; ++j) {
                const __bf16* bb = &vT[(16 * j + l15) * KT_STR + kb + lh * 16];
                scum[j] = wmma_bf16(af, ld16(bb, bb + 8), scum[j]);
            }
        }
        // emit scum^T (bf16) for q@scum
#pragma unroll
        for (int j = 0; j < 8; ++j)
#pragma unroll
            for (int r = 0; r < 8; ++r) {
                int d = 16 * w + r + 8 * lh;
                int e = 16 * j + l15;
                sT[e * ST_STR + d] = f2bf(scum[j][r]);
            }
        __syncthreads();

        // den = q . zcum
        {
            int i = tid & 63, qq = tid >> 6;
            float s = 0.0f;
#pragma unroll 8
            for (int d = qq * 32; d < qq * 32 + 32; ++d)
                s += bf2f(qS[i * QS_STR + d]) * zc[d];
            dp[i * 4 + qq] = s;
        }
        __syncthreads();
        if (tid < 64)
            dn[tid] = dp[tid * 4] + dp[tid * 4 + 1] + dp[tid * 4 + 2] + dp[tid * 4 + 3] + 1e-6f;

        // num = q @ scum  (M=i 64, K=d 128, N=e 128)
        v8f num[4];
#pragma unroll
        for (int jj = 0; jj < 4; ++jj) num[jj] = zero8();
#pragma unroll
        for (int s = 0; s < 4; ++s) {
            int kb = 32 * s;
            const __bf16* ab = &qS[(16 * rw + l15) * QS_STR + kb];
            v16bf af = ld16(ab + lh * 16, ab + lh * 16 + 8);
#pragma unroll
            for (int jj = 0; jj < 4; ++jj) {
                const __bf16* bb = &sT[(16 * (jt0 + jj) + l15) * ST_STR + kb + lh * 16];
                num[jj] = wmma_bf16(af, ld16(bb, bb + 8), num[jj]);
            }
        }
        __syncthreads();  // dn visible

        // out = g * num / den
#pragma unroll
        for (int jj = 0; jj < 4; ++jj)
#pragma unroll
            for (int r = 0; r < 8; ++r) {
                int i = 16 * rw + r + 8 * lh;
                int e = 16 * (jt0 + jj) + l15;
                size_t go = (rowbase + i) * D_MODEL + colbase + e;
                O[go] = f2bf(G[go] * num[jj][r] / dn[i]);
            }
        __syncthreads();  // before next chunk's staging
    }
}

// ---------------------------------------------------------------- launch
extern "C" void kernel_launch(void* const* d_in, const int* in_sizes, int n_in,
                              void* d_out, int out_size, void* d_ws, size_t ws_size,
                              hipStream_t stream) {
    const float* x  = (const float*)d_in[0];
    const float* qw = (const float*)d_in[1];
    const float* qb = (const float*)d_in[2];
    const float* kw = (const float*)d_in[3];
    const float* kb = (const float*)d_in[4];
    const float* vw = (const float*)d_in[5];
    const float* vb = (const float*)d_in[6];
    const float* gw = (const float*)d_in[7];
    const float* gb = (const float*)d_in[8];
    const float* ow = (const float*)d_in[9];
    const float* ob = (const float*)d_in[10];

    char* ws = (char*)d_ws;
    size_t off = 0;
    auto carve = [&](size_t bytes) -> char* {
        char* p = ws + off;
        off += (bytes + 255) & ~(size_t)255;
        return p;
    };
    const size_t NX = (size_t)M_ROWS * D_MODEL;
    const size_t NW = (size_t)D_MODEL * D_MODEL;
    __bf16* xb  = (__bf16*)carve(NX * 2);
    __bf16* qwb = (__bf16*)carve(NW * 2);
    __bf16* kwb = (__bf16*)carve(NW * 2);
    __bf16* vwb = (__bf16*)carve(NW * 2);
    __bf16* gwb = (__bf16*)carve(NW * 2);
    __bf16* owb = (__bf16*)carve(NW * 2);
    __bf16* qr  = (__bf16*)carve(NX * 2);  // elu(rope(q))+1
    __bf16* kr  = (__bf16*)carve(NX * 2);
    __bf16* vo  = (__bf16*)carve(NX * 2);
    float*  gf  = (float*)carve(NX * 4);   // sigmoid(g)
    __bf16* ao  = (__bf16*)carve(NX * 2);  // gated attention out

    cvt_f32_bf16<<<(int)(NX / 1024), 256, 0, stream>>>(x, xb, (int)NX);
    cvt_f32_bf16<<<(int)(NW / 1024), 256, 0, stream>>>(qw, qwb, (int)NW);
    cvt_f32_bf16<<<(int)(NW / 1024), 256, 0, stream>>>(kw, kwb, (int)NW);
    cvt_f32_bf16<<<(int)(NW / 1024), 256, 0, stream>>>(vw, vwb, (int)NW);
    cvt_f32_bf16<<<(int)(NW / 1024), 256, 0, stream>>>(gw, gwb, (int)NW);
    cvt_f32_bf16<<<(int)(NW / 1024), 256, 0, stream>>>(ow, owb, (int)NW);

    dim3 gg(D_MODEL / 128, M_ROWS / M_TILE);
    gemm_bf16<3><<<gg, 256, 0, stream>>>(xb, qwb, qb, (void*)qr);
    gemm_bf16<3><<<gg, 256, 0, stream>>>(xb, kwb, kb, (void*)kr);
    gemm_bf16<2><<<gg, 256, 0, stream>>>(xb, vwb, vb, (void*)vo);
    gemm_bf16<1><<<gg, 256, 0, stream>>>(xb, gwb, gb, (void*)gf);

    attn_kernel<<<B_SZ * NHEAD, 256, ATTN_SMEM, stream>>>(qr, kr, vo, gf, ao);

    gemm_bf16<0><<<gg, 256, 0, stream>>>(ao, owb, ob, d_out);
}
